// Attention_40973988004750
// MI455X (gfx1250) — matrix operand order repaired and checked by reading, not compiled, not run on hardware
//
#include <hip/hip_runtime.h>

typedef __attribute__((ext_vector_type(16))) _Float16 v16h;
typedef __attribute__((ext_vector_type(8)))  float    v8f;
typedef __attribute__((ext_vector_type(4)))  float    v4f;   // native vector for nontemporal store

#define B_ 2
#define H_ 16
#define S_ 2048
#define D_ 64
#define NEGBIG (-1e9f)
#define SCALE  (0.125f)          // 1/sqrt(64)

#define NW  8                    // waves per block
#define NT  256                  // threads per block
#define KPW (S_ / NW)            // 256 keys per wave in phase 1
#define VT_LD 72                 // padded LDS stride (halves) for transposed V tile

// ---- dynamic LDS layout (bytes) ----
#define SC_OFF   0                                 // float  sc [16][2048]    = 128 KB
#define PR_OFF   (SC_OFF   + 16 * S_ * 4)          // half   pr [16][2048]    =  64 KB
#define QA_OFF   (PR_OFF   + 16 * S_ * 2)          // half   qa [16][64]      =   2 KB
#define VT_OFF   (QA_OFF   + 16 * D_ * 2)          // half   vt [2][64][72]   =  18 KB (double buffer)
#define RED_OFF  (VT_OFF   + 2 * D_ * VT_LD * 2)   // float  red[16][16]      =   1 KB
#define RMAX_OFF (RED_OFF  + 16 * 16 * 4)          // float  rmax[16]
#define RSUM_OFF (RMAX_OFF + 16 * 4)               // float  rsum[16]
#define ACC_OFF  (RSUM_OFF + 16 * 4)               // float  accred[4][16][16] = 4 KB
#define SMEM_BYTES (ACC_OFF + 4 * 16 * 16 * 4)     // ~218 KB (< 320 KB/WGP)

union AF { v16h v; _Float16 h[16]; unsigned u[8]; };
union BF { v16h v; _Float16 h[16]; uint4 q4[2]; };
union H4 { _Float16 h[4]; unsigned long long ull; };

// half-pair offset inside a 16-bit A fragment: VGPR j, lane half `hi`
__device__ __forceinline__ int a_pair_off(int j, int hi) {
    return (hi ? 16 : 0) + 2 * j;
}

__global__ __launch_bounds__(NT)
void attn_fused_wmma(const float* __restrict__ q, const float* __restrict__ k,
                     const float* __restrict__ v, const float* __restrict__ mask,
                     float* __restrict__ outp, float* __restrict__ attnp)
{
    extern __shared__ char smem[];
    float*    sc     = (float*)   (smem + SC_OFF);
    _Float16* pr     = (_Float16*)(smem + PR_OFF);
    _Float16* qa     = (_Float16*)(smem + QA_OFF);
    float*    red    = (float*)   (smem + RED_OFF);
    float*    rmx    = (float*)   (smem + RMAX_OFF);
    float*    rsm    = (float*)   (smem + RSUM_OFF);
    float*    accred = (float*)   (smem + ACC_OFF);

    const int bh   = blockIdx.y;          // 0..31  (b*H + h)
    const int b    = bh / H_;
    const int q0   = blockIdx.x * 16;     // query-tile row base
    const int t    = threadIdx.x;
    const int lane = t & 31;
    const int w    = t >> 5;              // wave id 0..7
    const int n    = lane & 15;
    const int hi   = (lane >> 4) & 1;

    const float* qbase = q + (size_t)(bh * S_ + q0) * D_;
    const float* kbase = k + (size_t)bh * S_ * D_;
    const float* vbase = v + (size_t)bh * S_ * D_;

    // ---- stage Q tile (16x64 f32 -> f16 LDS), coalesced ----
    for (int f = t; f < 16 * D_; f += NT) qa[f] = (_Float16)qbase[f];
    __syncthreads();

    // ---- constant Q A-fragments for contraction K=0..31 and K=32..63 ----
    AF qf0, qf1;
#pragma unroll
    for (int j = 0; j < 8; ++j) {
        int off = a_pair_off(j, hi);
        qf0.u[j] = *(const unsigned*)&qa[n * D_ + off];
        qf1.u[j] = *(const unsigned*)&qa[n * D_ + 32 + off];
    }

    // ================= Phase 1: scores = Q K^T * scale + mask*NEG =================
    for (int kt = 0; kt < KPW / 16; ++kt) {
        const int kb = w * KPW + kt * 16;            // key-tile base for this wave
        const float* kr = kbase + (size_t)(kb + n) * D_;
        BF bf0, bf1;
#pragma unroll
        for (int c = 0; c < 4; ++c) {
            float4 a4 = ((const float4*)(kr + hi * 16))[c];        // d = hi*16 + 4c ..
            float4 b4 = ((const float4*)(kr + 32 + hi * 16))[c];   // d = 32 + hi*16 + 4c ..
            bf0.h[4*c+0] = (_Float16)a4.x; bf0.h[4*c+1] = (_Float16)a4.y;
            bf0.h[4*c+2] = (_Float16)a4.z; bf0.h[4*c+3] = (_Float16)a4.w;
            bf1.h[4*c+0] = (_Float16)b4.x; bf1.h[4*c+1] = (_Float16)b4.y;
            bf1.h[4*c+2] = (_Float16)b4.z; bf1.h[4*c+3] = (_Float16)b4.w;
        }
        v8f acc = {};
        acc = __builtin_amdgcn_wmma_f32_16x16x32_f16(false, qf0.v, false, bf0.v,
                                                     (short)0, acc, false, false);
        acc = __builtin_amdgcn_wmma_f32_16x16x32_f16(false, qf1.v, false, bf1.v,
                                                     (short)0, acc, false, false);
        const float mv = mask[b * S_ + kb + n] * NEGBIG;
#pragma unroll
        for (int r = 0; r < 8; ++r) {
            int M = r + hi * 8;                       // C/D layout: M = r + 8*hiLane
            sc[M * S_ + kb + n] = acc[r] * SCALE + mv;
        }
    }
    __syncthreads();

    // ================= Softmax over each of the 16 rows (2048 cols) =================
    {
        const int row = t & 15, slot = t >> 4;        // 16 partial slots per row
        // vectorized row max
        float m = -3.4e38f;
        for (int i = 0; i < S_ / (16 * 4); ++i) {     // 32 float4 chunks per slot
            float4 x = *(const float4*)&sc[row * S_ + (slot + 16 * i) * 4];
            m = fmaxf(m, fmaxf(fmaxf(x.x, x.y), fmaxf(x.z, x.w)));
        }
        red[row * 16 + slot] = m;
        __syncthreads();
        if (t < 16) {
            float mm = red[t * 16];
            for (int i = 1; i < 16; ++i) mm = fmaxf(mm, red[t * 16 + i]);
            rmx[t] = mm;
        }
        __syncthreads();
        // vectorized exp + sum, storing exp(s-max) back into sc
        const float rm = rmx[row];
        float s = 0.f;
        for (int i = 0; i < S_ / (16 * 4); ++i) {
            float* p4 = &sc[row * S_ + (slot + 16 * i) * 4];
            float4 x = *(const float4*)p4;
            x.x = __expf(x.x - rm); x.y = __expf(x.y - rm);
            x.z = __expf(x.z - rm); x.w = __expf(x.w - rm);
            *(float4*)p4 = x;
            s += (x.x + x.y) + (x.z + x.w);
        }
        red[row * 16 + slot] = s;
        __syncthreads();
        if (t < 16) {
            float ss = 0.f;
            for (int i = 0; i < 16; ++i) ss += red[t * 16 + i];
            rsm[t] = 1.0f / ss;
        }
        __syncthreads();
    }

    // probs -> global attn (fp32, non-temporal v4f, coalesced) and LDS pr (f16)
    float* attnrow = attnp + (size_t)(bh * S_ + q0) * S_;
    for (int f = t; f < 16 * (S_ / 4); f += NT) {      // 32 iterations
        int row = f >> 9, c4 = (f & 511) * 4;
        float4 x = *(const float4*)&sc[row * S_ + c4];
        const float is = rsm[row];
        v4f y;
        y.x = x.x * is; y.y = x.y * is; y.z = x.z * is; y.w = x.w * is;
        // non-temporal: do not let the 512MB attn stream evict K/V from L2
        __builtin_nontemporal_store(y, (v4f*)(attnrow + (size_t)row * S_ + c4));
        H4 h4;
        h4.h[0] = (_Float16)y.x; h4.h[1] = (_Float16)y.y;
        h4.h[2] = (_Float16)y.z; h4.h[3] = (_Float16)y.w;
        *(unsigned long long*)&pr[row * S_ + c4] = h4.ull;
    }
    __syncthreads();

    // ================= Phase 2: out = P @ V  (16 x 2048 . 2048 x 64) =================
    const int n0    = (w & 3) * 16;    // output d-tile for this wave
    const int khalf = w >> 2;          // which 32-key half of each staged 64-key tile
    v8f oacc = {};
    for (int it = 0; it < S_ / 64; ++it) {             // 32 iterations
        const int kk0 = it * 64;
        _Float16* vt = (_Float16*)(smem + VT_OFF) + (it & 1) * (D_ * VT_LD);

        // stage V[kk0:kk0+64][0:64] transposed into vt[d][key] (float4 global reads)
        for (int cch = t; cch < 64 * 16; cch += NT) {  // 4 float4 chunks per thread
            int key = cch >> 4, d4 = (cch & 15) * 4;
            float4 x = *(const float4*)&vbase[(size_t)(kk0 + key) * D_ + d4];
            vt[(d4 + 0) * VT_LD + key] = (_Float16)x.x;
            vt[(d4 + 1) * VT_LD + key] = (_Float16)x.y;
            vt[(d4 + 2) * VT_LD + key] = (_Float16)x.z;
            vt[(d4 + 3) * VT_LD + key] = (_Float16)x.w;
        }
        __syncthreads();   // single barrier per step: double-buffering handles WAR

        const int kb = kk0 + khalf * 32;
        AF af; BF bf;
#pragma unroll
        for (int j = 0; j < 8; ++j)
            af.u[j] = *(const unsigned*)&pr[n * S_ + kb + a_pair_off(j, hi)];
        const _Float16* vrow = vt + (n0 + n) * VT_LD + khalf * 32 + hi * 16;
        bf.q4[0] = ((const uint4*)vrow)[0];
        bf.q4[1] = ((const uint4*)vrow)[1];

        oacc = __builtin_amdgcn_wmma_f32_16x16x32_f16(false, af.v, false, bf.v,
                                                      (short)0, oacc, false, false);
    }

    // reduce the two k-halves and store out
    __syncthreads();
    if (khalf == 1) {
#pragma unroll
        for (int r = 0; r < 8; ++r)
            accred[((w & 3) * 16 + r + hi * 8) * 16 + n] = oacc[r];
    }
    __syncthreads();
    if (khalf == 0) {
        float* orow = outp + (size_t)(bh * S_ + q0) * D_;
#pragma unroll
        for (int r = 0; r < 8; ++r) {
            int M = r + hi * 8;
            orow[(size_t)M * D_ + n0 + n] = oacc[r] + accred[((w & 3) * 16 + M) * 16 + n];
        }
    }
}

extern "C" void kernel_launch(void* const* d_in, const int* in_sizes, int n_in,
                              void* d_out, int out_size, void* d_ws, size_t ws_size,
                              hipStream_t stream) {
    (void)in_sizes; (void)n_in; (void)out_size; (void)d_ws; (void)ws_size;
    const float* q    = (const float*)d_in[0];
    const float* k    = (const float*)d_in[1];
    const float* v    = (const float*)d_in[2];
    const float* mask = (const float*)d_in[3];
    float* outp  = (float*)d_out;                              // [B,H,S,D]
    float* attnp = outp + (size_t)B_ * H_ * S_ * D_;           // [B,H,S,S]

    static_assert(SMEM_BYTES <= 320 * 1024, "LDS budget exceeded");
    (void)hipFuncSetAttribute((const void*)attn_fused_wmma,
                              hipFuncAttributeMaxDynamicSharedMemorySize, SMEM_BYTES);

    dim3 grid(S_ / 16, B_ * H_);   // 128 q-tiles x 32 (b,h) = 4096 workgroups
    attn_fused_wmma<<<grid, NT, SMEM_BYTES, stream>>>(q, k, v, mask, outp, attnp);
}
